// LlamaAttention_23536420782093
// MI455X (gfx1250) — matrix operand order repaired and checked by reading, not compiled, not run on hardware
//
#include <hip/hip_runtime.h>
#include <hip/hip_bf16.h>
#include <cstdint>

// ---------------- problem constants ----------------
#define SL      1024          // sequence length
#define HIDD    2048          // hidden dim
#define NHEADS  16
#define KVHEADS 4
#define HDIM    128
#define GROUPS  4
#define FETCHMAX 204          // int(1024 * 0.2)
#define ALPHA_C 5.0f
#define NEGV    (-10000.0f)
#define SCALE_C 0.08838834764831845f   // 128^-0.5

typedef __bf16 bf16_t;
typedef __attribute__((ext_vector_type(16))) __bf16 v16bf;
typedef __attribute__((ext_vector_type(8)))  __bf16 bf16x8;
typedef __attribute__((ext_vector_type(8)))  float  v8f;

// ---------------- wave32 reductions ----------------
__device__ __forceinline__ float wredMaxF(float v) {
#pragma unroll
  for (int m = 16; m > 0; m >>= 1) v = fmaxf(v, __shfl_xor(v, m, 32));
  return v;
}
__device__ __forceinline__ float wredSumF(float v) {
#pragma unroll
  for (int m = 16; m > 0; m >>= 1) v += __shfl_xor(v, m, 32);
  return v;
}
__device__ __forceinline__ int wredSumI(int v) {
#pragma unroll
  for (int m = 16; m > 0; m >>= 1) v += __shfl_xor(v, m, 32);
  return v;
}

// ---------------- WMMA fragment loaders (ISA 7.12.2 layouts) ----------------
// A: 16x32 bf16. lane L: row m0+(L&15); K chunks [kc+half*8 .. +8) and [+16 .. +24)
__device__ __forceinline__ v16bf load_fragA(const bf16_t* base, int ld, int m0, int kc, int lane) {
  int r  = m0 + (lane & 15);
  int k0 = kc + ((lane >> 4) << 3);
  const bf16_t* p = base + (size_t)r * ld + k0;
  union { v16bf v; bf16x8 h[2]; } u;
  u.h[0] = *(const bf16x8*)(p);
  u.h[1] = *(const bf16x8*)(p + 16);
  return u.v;
}
// B: 32x16 bf16 (W row-major N x K, computing A @ W^T). lane L: W row n0+(L&15),
// K chunks [kc+half*8 .. +8) and [+16 .. +24)
__device__ __forceinline__ v16bf load_fragB(const bf16_t* base, int ld, int n0, int kc, int lane) {
  int n  = n0 + (lane & 15);
  int k0 = kc + ((lane >> 4) << 3);
  const bf16_t* p = base + (size_t)n * ld + k0;
  union { v16bf v; bf16x8 h[2]; } u;
  u.h[0] = *(const bf16x8*)(p);
  u.h[1] = *(const bf16x8*)(p + 16);
  return u.v;
}

// ---------------- generic batched NT GEMM: C = scale * A @ B^T (+ causal NEG) ----
// Wave tile: 32 (M) x 64 (N) -> 8 accumulators; B fragment reused by both M rows.
// per blockIdx.z: A += z*aStride, B += (z/bDiv)*bStride, C += z*cStride
__global__ void __launch_bounds__(256) wmma_gemm_nt(
    const bf16_t* __restrict__ A, const bf16_t* __restrict__ B, float* __restrict__ C,
    int M, int N, int Kd, int lda, int ldb, int ldc,
    long long aStride, long long bStride, int bDiv, long long cStride,
    float scale, int causal)
{
  int lane = threadIdx.x & 31;
  int wave = threadIdx.x >> 5;
  int z = blockIdx.z;
  A += (size_t)z * (size_t)aStride;
  B += (size_t)(z / bDiv) * (size_t)bStride;
  C += (size_t)z * (size_t)cStride;

  int tilesN = N >> 6;                       // 64-wide, 32-tall wave tiles
  int gid = blockIdx.x * 8 + wave;
  int total = (M >> 5) * tilesN;
  if (gid >= total) return;
  int tm = gid / tilesN, tn = gid - tm * tilesN;
  int m0 = tm << 5, n0 = tn << 6;

  v8f acc0[4] = {v8f{}, v8f{}, v8f{}, v8f{}};
  v8f acc1[4] = {v8f{}, v8f{}, v8f{}, v8f{}};
#pragma unroll 2
  for (int kc = 0; kc < Kd; kc += 32) {
    v16bf a0 = load_fragA(A, lda, m0,      kc, lane);
    v16bf a1 = load_fragA(A, lda, m0 + 16, kc, lane);
#pragma unroll
    for (int t = 0; t < 4; ++t) {
      v16bf b = load_fragB(B, ldb, n0 + t * 16, kc, lane);
      acc0[t] = __builtin_amdgcn_wmma_f32_16x16x32_bf16(
          false, a0, false, b, (short)0, acc0[t], false, false);
      acc1[t] = __builtin_amdgcn_wmma_f32_16x16x32_bf16(
          false, a1, false, b, (short)0, acc1[t], false, false);
    }
  }
  int half = lane >> 4;
  int col  = lane & 15;
#pragma unroll
  for (int t = 0; t < 4; ++t) {
    int n = n0 + t * 16 + col;
#pragma unroll
    for (int r = 0; r < 8; ++r) {
      int m = m0 + r + half * 8;
      float v0 = acc0[t][r] * scale;
      float v1 = acc1[t][r] * scale;
      if (causal) {
        if (n > m) v0 += NEGV;
        if (n > m + 16) v1 += NEGV;
      }
      C[(size_t)m * ldc + n] = v0;
      C[(size_t)(m + 16) * ldc + n] = v1;
    }
  }
}

// ---------------- elementwise kernels ----------------
__global__ void cvt_f32_bf16(const float* __restrict__ x, bf16_t* __restrict__ y, int n) {
  int i = blockIdx.x * blockDim.x + threadIdx.x;
  if (i < n) y[i] = (bf16_t)x[i];
}

__global__ void zero_i32(int* p, int n) {
  int i = blockIdx.x * blockDim.x + threadIdx.x;
  if (i < n) p[i] = 0;
}

// RoPE on row-major [SL][nhd] f32 -> bf16, cos/sin are [SL][HDIM]
__global__ void rope_kernel(const float* __restrict__ X, const float* __restrict__ cosT,
                            const float* __restrict__ sinT, bf16_t* __restrict__ out, int nhd) {
  int idx = blockIdx.x * blockDim.x + threadIdx.x;
  int total = SL * nhd;
  if (idx >= total) return;
  int s = idx / nhd;
  int c = idx - s * nhd;
  int d = c & (HDIM - 1);
  float x = X[idx];
  float other = (d < HDIM / 2) ? -X[idx + HDIM / 2] : X[idx - HDIM / 2];
  float r = x * cosT[s * HDIM + d] + other * sinT[s * HDIM + d];
  out[idx] = (bf16_t)r;
}

// V f32 [SL][KVHEADS*HDIM] -> Vt bf16 [KVHEADS][HDIM][SL]
__global__ void vtrans_kernel(const float* __restrict__ V, bf16_t* __restrict__ Vt) {
  int idx = blockIdx.x * blockDim.x + threadIdx.x;
  int total = KVHEADS * HDIM * SL;
  if (idx >= total) return;
  int kv = idx / (HDIM * SL);
  int rem = idx - kv * (HDIM * SL);
  int d = rem / SL;
  int s = rem - d * SL;
  Vt[idx] = (bf16_t)V[(size_t)s * (KVHEADS * HDIM) + kv * HDIM + d];
}

// ---------------- mask statistics: per (head,row) max & near-max count ----------
__global__ void __launch_bounds__(256) stats_kernel(const float* __restrict__ spec, int* __restrict__ cnt) {
  int lane = threadIdx.x & 31;
  int gid = blockIdx.x * 8 + (threadIdx.x >> 5);
  int h = gid >> 10, i = gid & (SL - 1);
  const float* row = spec + ((size_t)h * SL + i) * SL;
  float va[32];
  float mx = -3.0e38f;
#pragma unroll
  for (int t = 0; t < 32; ++t) { va[t] = row[lane + t * 32]; mx = fmaxf(mx, va[t]); }
  mx = wredMaxF(mx);
  float thr = mx - ALPHA_C;
  int c = 0;
#pragma unroll
  for (int t = 0; t < 32; ++t) c += (va[t] >= thr) ? 1 : 0;
  c = wredSumI(c);
  if (lane == 0) atomicAdd(&cnt[i], c);
}

__global__ void fn_kernel(const int* __restrict__ cnt, int* __restrict__ fn) {
  int i = blockIdx.x * blockDim.x + threadIdx.x;
  if (i < SL) {
    int c = cnt[i] / NHEADS;         // trunc(mean) for positive counts
    fn[i] = (c < FETCHMAX) ? c : FETCHMAX;
  }
}

// ---------------- exact fn-th largest via 32-step bit binary search -------------
__global__ void __launch_bounds__(256) thresh_kernel(const float* __restrict__ spec,
                                                     const int* __restrict__ fn,
                                                     float* __restrict__ thr) {
  int lane = threadIdx.x & 31;
  int gid = blockIdx.x * 8 + (threadIdx.x >> 5);
  int h = gid >> 10, i = gid & (SL - 1);
  int f = fn[i];
  if (f <= 0) { if (lane == 0) thr[h * SL + i] = 3.0e38f; return; }
  const float* row = spec + ((size_t)h * SL + i) * SL;
  unsigned ka[32];
#pragma unroll
  for (int t = 0; t < 32; ++t) {
    unsigned b = __float_as_uint(row[lane + t * 32]);
    ka[t] = (b & 0x80000000u) ? ~b : (b | 0x80000000u);   // order-preserving map
  }
  unsigned lo = 0u, hi = 0xFFFFFFFFu;
  while (lo < hi) {                     // find max u with count(key >= u) >= f
    unsigned d = hi - lo;
    unsigned mid = lo + (d >> 1) + (d & 1u);   // upper mid, overflow-safe
    int c = 0;
#pragma unroll
    for (int t = 0; t < 32; ++t) c += (ka[t] >= mid) ? 1 : 0;
    c = wredSumI(c);
    if (c >= f) lo = mid; else hi = mid - 1;
  }
  unsigned b = (lo & 0x80000000u) ? (lo & 0x7FFFFFFFu) : ~lo;
  if (lane == 0) thr[h * SL + i] = __uint_as_float(b);
}

// ---------------- fused fetch-mask + softmax -> bf16 probabilities --------------
__global__ void __launch_bounds__(256) softmax_mask_kernel(
    const float* __restrict__ attn, const float* __restrict__ spec,
    const float* __restrict__ thr, const int* __restrict__ fn, bf16_t* __restrict__ P) {
  int lane = threadIdx.x & 31;
  int gid = blockIdx.x * 8 + (threadIdx.x >> 5);
  int h = gid >> 10, i = gid & (SL - 1);
  const float* srow = attn + ((size_t)h * SL + i) * SL;   // qk*SCALE + causal NEG
  const float* arow = spec + ((size_t)h * SL + i) * SL;   // spec*SCALE + causal NEG
  float tv = thr[h * SL + i];
  int f = fn[i];
  bool full = (i < FETCHMAX);
  float vals[32];
  float mx = -3.0e38f;
#pragma unroll
  for (int t = 0; t < 32; ++t) {
    int j = lane + t * 32;
    bool causalOk = (j <= i);
    bool fetch = full ? causalOk : (causalOk && (f > 0) && (arow[j] >= tv));
    float v = srow[j] + (fetch ? 0.0f : NEGV);
    vals[t] = v;
    mx = fmaxf(mx, v);
  }
  mx = wredMaxF(mx);
  float s = 0.0f;
#pragma unroll
  for (int t = 0; t < 32; ++t) { float e = __expf(vals[t] - mx); vals[t] = e; s += e; }
  s = wredSumF(s);
  float inv = 1.0f / s;
  bf16_t* prow = P + ((size_t)h * SL + i) * SL;
#pragma unroll
  for (int t = 0; t < 32; ++t) prow[lane + t * 32] = (bf16_t)(vals[t] * inv);
}

// ---------------- host side ----------------
static inline dim3 grid1(size_t n, int blk = 256) { return dim3((unsigned)((n + blk - 1) / blk)); }
static inline unsigned gemmBlocks(int M, int N) {
  unsigned waves = (unsigned)((M >> 5) * (N >> 6));
  return (waves + 7) / 8;
}

extern "C" void kernel_launch(void* const* d_in, const int* in_sizes, int n_in,
                              void* d_out, int out_size, void* d_ws, size_t ws_size,
                              hipStream_t stream) {
  (void)in_sizes; (void)n_in; (void)out_size; (void)ws_size;
  const float* hidden = (const float*)d_in[0];
  const float* prev   = (const float*)d_in[1];
  const float* cosT   = (const float*)d_in[2];
  const float* sinT   = (const float*)d_in[3];
  const float* Wq     = (const float*)d_in[4];
  const float* Wk     = (const float*)d_in[5];
  const float* Wv     = (const float*)d_in[6];
  const float* Wo     = (const float*)d_in[7];
  float* out = (float*)d_out;

  char* ws = (char*)d_ws;
  size_t off = 0;
  auto alloc = [&](size_t bytes) -> void* {
    void* p = ws + off;
    off = (off + bytes + 255) & ~(size_t)255;
    return p;
  };
  const size_t nHid = (size_t)SL * HIDD;            // 2M
  const size_t nKv  = (size_t)SL * KVHEADS * HDIM;  // 0.5M
  const size_t nWq  = (size_t)NHEADS * HDIM * HIDD; // 4.19M
  const size_t nWkv = (size_t)KVHEADS * HDIM * HIDD;
  const size_t nSS  = (size_t)NHEADS * SL * SL;     // 16M

  bf16_t* hid_bf = (bf16_t*)alloc(nHid * 2);
  bf16_t* prv_bf = (bf16_t*)alloc(nHid * 2);
  bf16_t* Wq_bf  = (bf16_t*)alloc(nWq * 2);
  bf16_t* Wk_bf  = (bf16_t*)alloc(nWkv * 2);
  bf16_t* Wv_bf  = (bf16_t*)alloc(nWkv * 2);
  bf16_t* Wo_bf  = (bf16_t*)alloc(nWq * 2);
  float*  Qf     = (float*)alloc(nHid * 4);
  float*  Kf     = (float*)alloc(nKv * 4);
  float*  Vf     = (float*)alloc(nKv * 4);
  float*  QSf    = (float*)alloc(nHid * 4);
  bf16_t* Q_bf   = (bf16_t*)alloc(nHid * 2);
  bf16_t* K_bf   = (bf16_t*)alloc(nKv * 2);
  bf16_t* QS_bf  = (bf16_t*)alloc(nHid * 2);
  bf16_t* Vt_bf  = (bf16_t*)alloc(nKv * 2);
  float*  spec   = (float*)alloc(nSS * 4);
  float*  attn   = (float*)alloc(nSS * 4);
  bf16_t* P_bf   = (bf16_t*)alloc(nSS * 2);
  float*  ctx    = (float*)alloc(nHid * 4);
  bf16_t* ctx_bf = (bf16_t*)alloc(nHid * 2);
  int*    cnt    = (int*)alloc(SL * 4);
  int*    fnb    = (int*)alloc(SL * 4);
  float*  thr    = (float*)alloc((size_t)NHEADS * SL * 4);

  // 1) bf16 conversions of activations and weights
  cvt_f32_bf16<<<grid1(nHid), 256, 0, stream>>>(hidden, hid_bf, (int)nHid);
  cvt_f32_bf16<<<grid1(nHid), 256, 0, stream>>>(prev, prv_bf, (int)nHid);
  cvt_f32_bf16<<<grid1(nWq), 256, 0, stream>>>(Wq, Wq_bf, (int)nWq);
  cvt_f32_bf16<<<grid1(nWkv), 256, 0, stream>>>(Wk, Wk_bf, (int)nWkv);
  cvt_f32_bf16<<<grid1(nWkv), 256, 0, stream>>>(Wv, Wv_bf, (int)nWkv);
  cvt_f32_bf16<<<grid1(nWq), 256, 0, stream>>>(Wo, Wo_bf, (int)nWq);

  // 2) projections (WMMA):  X @ W^T
  wmma_gemm_nt<<<dim3(gemmBlocks(SL, HIDD), 1, 1), 256, 0, stream>>>(hid_bf, Wq_bf, Qf,
      SL, HIDD, HIDD, HIDD, HIDD, HIDD, 0, 0, 1, 0, 1.0f, 0);
  wmma_gemm_nt<<<dim3(gemmBlocks(SL, KVHEADS * HDIM), 1, 1), 256, 0, stream>>>(hid_bf, Wk_bf, Kf,
      SL, KVHEADS * HDIM, HIDD, HIDD, HIDD, KVHEADS * HDIM, 0, 0, 1, 0, 1.0f, 0);
  wmma_gemm_nt<<<dim3(gemmBlocks(SL, KVHEADS * HDIM), 1, 1), 256, 0, stream>>>(hid_bf, Wv_bf, Vf,
      SL, KVHEADS * HDIM, HIDD, HIDD, HIDD, KVHEADS * HDIM, 0, 0, 1, 0, 1.0f, 0);
  wmma_gemm_nt<<<dim3(gemmBlocks(SL, HIDD), 1, 1), 256, 0, stream>>>(prv_bf, Wq_bf, QSf,
      SL, HIDD, HIDD, HIDD, HIDD, HIDD, 0, 0, 1, 0, 1.0f, 0);

  // 3) RoPE (-> bf16) and V transpose (-> bf16 [kv][d][s])
  rope_kernel<<<grid1(nHid), 256, 0, stream>>>(Qf, cosT, sinT, Q_bf, HIDD);
  rope_kernel<<<grid1(nHid), 256, 0, stream>>>(QSf, cosT, sinT, QS_bf, HIDD);
  rope_kernel<<<grid1(nKv), 256, 0, stream>>>(Kf, cosT, sinT, K_bf, KVHEADS * HDIM);
  vtrans_kernel<<<grid1(nKv), 256, 0, stream>>>(Vf, Vt_bf);

  // 4) spec & real attention scores per head (WMMA, causal NEG folded in)
  wmma_gemm_nt<<<dim3(gemmBlocks(SL, SL), 1, NHEADS), 256, 0, stream>>>(QS_bf, K_bf, spec,
      SL, SL, HDIM, HIDD, KVHEADS * HDIM, SL,
      HDIM, HDIM, GROUPS, (long long)SL * SL, SCALE_C, 1);
  wmma_gemm_nt<<<dim3(gemmBlocks(SL, SL), 1, NHEADS), 256, 0, stream>>>(Q_bf, K_bf, attn,
      SL, SL, HDIM, HIDD, KVHEADS * HDIM, SL,
      HDIM, HDIM, GROUPS, (long long)SL * SL, SCALE_C, 1);

  // 5) mask statistics -> fn -> per (head,row) exact threshold
  zero_i32<<<grid1(SL), 256, 0, stream>>>(cnt, SL);
  stats_kernel<<<dim3(2048), 256, 0, stream>>>(spec, cnt);
  fn_kernel<<<grid1(SL), 256, 0, stream>>>(cnt, fnb);
  thresh_kernel<<<dim3(2048), 256, 0, stream>>>(spec, fnb, thr);

  // 6) fused fetch-mask + softmax -> bf16 probabilities
  softmax_mask_kernel<<<dim3(2048), 256, 0, stream>>>(attn, spec, thr, fnb, P_bf);

  // 7) context = P @ V (WMMA, per head, Vt gives contiguous B frags)
  wmma_gemm_nt<<<dim3(gemmBlocks(SL, HDIM), 1, NHEADS), 256, 0, stream>>>(P_bf, Vt_bf, ctx,
      SL, HDIM, SL, SL, SL, HIDD,
      (long long)SL * SL, (long long)HDIM * SL, GROUPS, HDIM, 1.0f, 0);

  // 8) output projection (WMMA)
  cvt_f32_bf16<<<grid1(nHid), 256, 0, stream>>>(ctx, ctx_bf, (int)nHid);
  wmma_gemm_nt<<<dim3(gemmBlocks(SL, HIDD), 1, 1), 256, 0, stream>>>(ctx_bf, Wo_bf, out,
      SL, HIDD, HIDD, HIDD, HIDD, HIDD, 0, 0, 1, 0, 1.0f, 0);
}
